// CBlock2_49349174231388
// MI455X (gfx1250) — matrix operand order repaired and checked by reading, not compiled, not run on hardware
//
#include <hip/hip_runtime.h>
#include <math.h>

// ---------------- problem constants ----------------
#define Bq   8
#define Nq   1024
#define Cq   512
#define Hq   8
#define HDq  64
#define HIDq 2048
#define LN_EPS  1e-5f
#define ATT_EPS 1e-6f
#define SCALE   0.125f            // HD^-0.5
#define EPS_N   (ATT_EPS / (float)Nq)

typedef __attribute__((ext_vector_type(16))) __bf16 v16bf;
typedef __attribute__((ext_vector_type(8)))  float  v8f;

union FragBF { v16bf v; uint4 q[2]; };

static __device__ __forceinline__ v8f wmma_bf16(v16bf a, v16bf b, v8f c) {
  return __builtin_amdgcn_wmma_f32_16x16x32_bf16(false, a, false, b, (short)0, c,
                                                 false, false);
}

// A fragment: 16x32 bf16, row-major source with leading dim ld.
// lane holds row (lane&15); K offset 8*(lane>=16); dwords 0-3: K koff..koff+7,
// dwords 4-7: K 16+koff..16+koff+7  -> two 16-byte loads.
static __device__ __forceinline__ v16bf load_fragA(const __bf16* base, int row0,
                                                   int ld, int kk, unsigned lane) {
  unsigned r = lane & 15u;
  unsigned koff = (lane >> 4) * 8u;
  const __bf16* p = base + (size_t)(row0 + r) * (size_t)ld + kk + koff;
  FragBF f;
  f.q[0] = *(const uint4*)p;
  f.q[1] = *(const uint4*)(p + 16);
  return f.v;
}

// B fragment: 32x16 bf16 where column n = row (col0+n) of a row-major matrix
// (i.e. computes A * W^T).  lane holds column (lane&15); K offset 8*(lane>=16);
// dwords 0-3: K kb..kb+7, dwords 4-7: K 16+kb..16+kb+7 -> two 16-byte loads.
static __device__ __forceinline__ v16bf load_fragB(const __bf16* Wb, int col0,
                                                   int ld, int kk, unsigned lane) {
  unsigned c = lane & 15u;
  unsigned kb = (lane >> 4) * 8u;
  const __bf16* p = Wb + (size_t)(col0 + c) * (size_t)ld + kk + kb;
  FragBF f;
  f.q[0] = *(const uint4*)p;
  f.q[1] = *(const uint4*)(p + 16);
  return f.v;
}

// A fragment from an LDS 16x32 tile with row stride 40 bf16 (80B, conflict pad).
static __device__ __forceinline__ v16bf load_fragA_lds(const __bf16* tile,
                                                       unsigned lane) {
  unsigned r = lane & 15u;
  unsigned koff = (lane >> 4) * 8u;
  const __bf16* p = tile + r * 40u + koff;
  FragBF f;
  f.q[0] = *(const uint4*)p;
  f.q[1] = *(const uint4*)(p + 16);
  return f.v;
}

// ---------------- fp32 -> bf16 convert ----------------
__global__ void cvt_bf16_kernel(const float* __restrict__ in,
                                __bf16* __restrict__ out, int n) {
  for (int i = blockIdx.x * blockDim.x + threadIdx.x; i < n;
       i += gridDim.x * blockDim.x)
    out[i] = (__bf16)in[i];
}

// ---------------- LayerNorm (one wave per 512-wide row) ----------------
__global__ __launch_bounds__(128) void ln_kernel(const float* __restrict__ x,
                                                 const float* __restrict__ g,
                                                 const float* __restrict__ bt,
                                                 __bf16* __restrict__ out) {
  unsigned lane = threadIdx.x & 31u;
  unsigned wid = threadIdx.x >> 5;
  int row = blockIdx.x * 4 + wid;
  const float* p = x + (size_t)row * Cq;
  float v[16], s = 0.f;
#pragma unroll
  for (int i = 0; i < 16; ++i) { v[i] = p[lane + 32 * i]; s += v[i]; }
#pragma unroll
  for (int m = 16; m >= 1; m >>= 1) s += __shfl_xor(s, m, 32);
  float mu = s * (1.0f / Cq);
  float sq = 0.f;
#pragma unroll
  for (int i = 0; i < 16; ++i) { float d = v[i] - mu; sq += d * d; }
#pragma unroll
  for (int m = 16; m >= 1; m >>= 1) sq += __shfl_xor(sq, m, 32);
  float rs = rsqrtf(sq * (1.0f / Cq) + LN_EPS);
  __bf16* o = out + (size_t)row * Cq;
#pragma unroll
  for (int i = 0; i < 16; ++i) {
    int c = lane + 32 * i;
    o[c] = (__bf16)((v[i] - mu) * rs * g[c] + bt[c]);
  }
}

// ---------------- QKV projection (WMMA, permuted stores) ----------------
// which: 0 -> q (b,h,n,d) row-major, 1 -> k (b,h,n,d), 2 -> v transposed (b,h,d,n)
__global__ __launch_bounds__(128) void qkv_gemm_kernel(
    const __bf16* __restrict__ A, const __bf16* __restrict__ W,
    __bf16* __restrict__ out, int which) {
  unsigned lane = threadIdx.x & 31u;
  unsigned wid = threadIdx.x >> 5;
  int tile = blockIdx.x * 4 + wid;
  const int tiles_n = Cq / 16;
  int tm = tile / tiles_n, tn = tile % tiles_n;
  v8f acc = {};
  for (int kk = 0; kk < Cq; kk += 32) {
    v16bf a = load_fragA(A, tm * 16, Cq, kk, lane);
    v16bf b = load_fragB(W, tn * 16, Cq, kk, lane);
    acc = wmma_bf16(a, b, acc);
  }
  unsigned col = lane & 15u;
  unsigned roff = (lane >> 4) * 8u;
  int c = tn * 16 + col, h = c >> 6, d = c & 63;
#pragma unroll
  for (int j = 0; j < 8; ++j) {
    int row = tm * 16 + j + roff;
    int b0 = row >> 10, n = row & 1023;
    size_t idx;
    if (which < 2) idx = (((size_t)(b0 * Hq + h) * Nq) + n) * HDq + d;
    else           idx = (((size_t)(b0 * Hq + h) * HDq) + d) * Nq + n;
    out[idx] = (__bf16)acc[j];
  }
}

// ---------------- per-(b,h) sum of V rows ----------------
__global__ void vsum_kernel(const __bf16* __restrict__ vT,
                            float* __restrict__ vsum) {
  int bh = blockIdx.x, d = threadIdx.x;
  const __bf16* p = vT + ((size_t)bh * HDq + d) * Nq;
  float s = 0.f;
  for (int n = 0; n < Nq; ++n) s += (float)p[n];
  vsum[bh * HDq + d] = s;
}

// ---------------- clustered attention (flash-style, one 16-query tile/wave) ---
__global__ __launch_bounds__(128) void attn_kernel(
    const __bf16* __restrict__ qb, const __bf16* __restrict__ kb,
    const __bf16* __restrict__ vT, const float* __restrict__ vsum,
    const int* __restrict__ idxc, __bf16* __restrict__ attn_o) {
  __shared__ __attribute__((aligned(16))) __bf16 sP[4][16 * 40];
  unsigned lane = threadIdx.x & 31u;
  unsigned wid = threadIdx.x >> 5;
  unsigned gw = blockIdx.x * 4 + wid;
  unsigned bh = gw >> 6;           // 64 query tiles per (b,h)
  unsigned qt = gw & 63u;
  unsigned b = bh >> 3, h = bh & 7u;
  const __bf16* Q  = qb + (size_t)bh * Nq * HDq;
  const __bf16* Km = kb + (size_t)bh * Nq * HDq;
  const __bf16* Vt = vT + (size_t)bh * HDq * Nq;
  int qbase = qt * 16;
  unsigned col = lane & 15u;
  unsigned roff = (lane >> 4) * 8u;

  v16bf qa0 = load_fragA(Q, qbase, HDq, 0, lane);
  v16bf qa1 = load_fragA(Q, qbase, HDq, 32, lane);

  int qc[8];
#pragma unroll
  for (int j = 0; j < 8; ++j) qc[j] = idxc[b * Nq + qbase + j + roff];

  v8f o0 = {}, o1 = {}, o2 = {}, o3 = {};
  float den[8];
#pragma unroll
  for (int j = 0; j < 8; ++j) den[j] = 0.f;
  __bf16* myP = sP[wid];

  for (int jb = 0; jb < Nq; jb += 32) {
    // scores: two 16x16 tiles over a 32-key chunk, K = HD = 64
    v8f s0 = {}, s1 = {};
    s0 = wmma_bf16(qa0, load_fragB(Km, jb, HDq, 0, lane), s0);
    s0 = wmma_bf16(qa1, load_fragB(Km, jb, HDq, 32, lane), s0);
    s1 = wmma_bf16(qa0, load_fragB(Km, jb + 16, HDq, 0, lane), s1);
    s1 = wmma_bf16(qa1, load_fragB(Km, jb + 16, HDq, 32, lane), s1);

    int kc0 = idxc[b * Nq + jb + col];
    int kc1 = idxc[b * Nq + jb + 16 + col];
    float p0[8], p1[8];
#pragma unroll
    for (int j = 0; j < 8; ++j) {
      p0[j] = (qc[j] == kc0) ? __expf(s0[j] * SCALE) : 0.f;
      p1[j] = (qc[j] == kc1) ? __expf(s1[j] * SCALE) : 0.f;
    }
    // denominator: row sums within each 16-lane half (rows j / j+8)
#pragma unroll
    for (int j = 0; j < 8; ++j) {
      float r = p0[j] + p1[j];
      r += __shfl_xor(r, 1, 16);
      r += __shfl_xor(r, 2, 16);
      r += __shfl_xor(r, 4, 16);
      r += __shfl_xor(r, 8, 16);
      den[j] += r;
    }
    // D-layout -> A-layout re-stripe through wave-private LDS (DS in-order)
#pragma unroll
    for (int j = 0; j < 8; ++j) {
      unsigned m = j + roff;
      myP[m * 40 + col] = (__bf16)p0[j];
      myP[m * 40 + 16 + col] = (__bf16)p1[j];
    }
    asm volatile("" ::: "memory");
    v16bf pa = load_fragA_lds(myP, lane);
    asm volatile("" ::: "memory");

    // out += P(16x32) x V(32x64): 4 output tiles, V pre-transposed (d-major)
    o0 = wmma_bf16(pa, load_fragB(Vt, 0, Nq, jb, lane), o0);
    o1 = wmma_bf16(pa, load_fragB(Vt, 16, Nq, jb, lane), o1);
    o2 = wmma_bf16(pa, load_fragB(Vt, 32, Nq, jb, lane), o2);
    o3 = wmma_bf16(pa, load_fragB(Vt, 48, Nq, jb, lane), o3);
  }

  // out = (acc + (eps/N)*sum_j v_j) / (den + eps)
  v8f oArr[4] = {o0, o1, o2, o3};
#pragma unroll
  for (int t = 0; t < 4; ++t) {
    float vs = vsum[bh * HDq + t * 16 + col];
#pragma unroll
    for (int j = 0; j < 8; ++j) {
      unsigned m = j + roff;
      float val = (oArr[t][j] + EPS_N * vs) / (den[j] + ATT_EPS);
      attn_o[((size_t)(b * Nq + qbase + m)) * Cq + h * HDq + t * 16 + col] =
          (__bf16)val;
    }
  }
}

// ---------------- generic WMMA GEMM: out = epilogue(A * W^T + bias) ---------
// mode 1: f32 out = acc + bias + resid     (proj / mlp2 with residual)
// mode 2: bf16 out = gelu(acc + bias)      (mlp1)
__global__ __launch_bounds__(128) void gemm_kernel(
    const __bf16* __restrict__ A, const __bf16* __restrict__ W,
    const float* __restrict__ bias, const float* __restrict__ resid,
    float* __restrict__ outF, __bf16* __restrict__ outH, int M, int K, int Nout,
    int mode) {
  unsigned lane = threadIdx.x & 31u;
  unsigned wid = threadIdx.x >> 5;
  int tile = blockIdx.x * 4 + wid;
  int tiles_n = Nout / 16;
  if (tile >= (M / 16) * tiles_n) return;
  int tm = tile / tiles_n, tn = tile % tiles_n;
  v8f acc = {};
  unsigned r = lane & 15u, koff = (lane >> 4) * 8u, kb = (lane >> 4) * 8u;
  const __bf16* pa = A + (size_t)(tm * 16 + r) * K + koff;
  const __bf16* pb = W + (size_t)(tn * 16 + r) * K + kb;
  for (int kk = 0; kk < K; kk += 32) {
    __builtin_prefetch(pa + kk + 64, 0, 0);
    __builtin_prefetch(pb + kk + 64, 0, 0);
    FragBF fa, fb;
    fa.q[0] = *(const uint4*)(pa + kk);
    fa.q[1] = *(const uint4*)(pa + kk + 16);
    fb.q[0] = *(const uint4*)(pb + kk);
    fb.q[1] = *(const uint4*)(pb + kk + 16);
    acc = wmma_bf16(fa.v, fb.v, acc);
  }
  unsigned col = lane & 15u;
  unsigned roff = (lane >> 4) * 8u;
  float bv = bias ? bias[tn * 16 + col] : 0.f;
#pragma unroll
  for (int j = 0; j < 8; ++j) {
    size_t idx = (size_t)(tm * 16 + j + roff) * Nout + tn * 16 + col;
    float v = acc[j] + bv;
    if (mode == 1) {
      outF[idx] = v + (resid ? resid[idx] : 0.f);
    } else {
      float gl = 0.5f * v * (1.0f + erff(v * 0.70710678118654752f));
      outH[idx] = (__bf16)gl;
    }
  }
}

// ---------------- host-side orchestration ----------------
extern "C" void kernel_launch(void* const* d_in, const int* in_sizes, int n_in,
                              void* d_out, int out_size, void* d_ws,
                              size_t ws_size, hipStream_t stream) {
  const float* x_token = (const float*)d_in[0];
  // d_in[1] = x_path (unused by reference)
  const float* wq = (const float*)d_in[2];
  const float* wk = (const float*)d_in[3];
  const float* wv = (const float*)d_in[4];
  const float* w_proj = (const float*)d_in[5];
  const float* b_proj = (const float*)d_in[6];
  const float* g1 = (const float*)d_in[7];
  const float* b1 = (const float*)d_in[8];
  const float* g2 = (const float*)d_in[9];
  const float* b2 = (const float*)d_in[10];
  const float* w1 = (const float*)d_in[11];
  const float* bb1 = (const float*)d_in[12];
  const float* w2 = (const float*)d_in[13];
  const float* bb2 = (const float*)d_in[14];
  const int* idxc = (const int*)d_in[15];

  char* ws = (char*)d_ws;
  size_t off = 0;
  auto alloc = [&](size_t bytes) {
    char* p = ws + off;
    off += (bytes + 255) & ~(size_t)255;
    return p;
  };
  const size_t BNC = (size_t)Bq * Nq * Cq;
  __bf16* tb   = (__bf16*)alloc(BNC * 2);              // LN1(x) bf16
  __bf16* wqb  = (__bf16*)alloc((size_t)Cq * Cq * 2);
  __bf16* wkb  = (__bf16*)alloc((size_t)Cq * Cq * 2);
  __bf16* wvb  = (__bf16*)alloc((size_t)Cq * Cq * 2);
  __bf16* wpb  = (__bf16*)alloc((size_t)Cq * Cq * 2);
  __bf16* w1b  = (__bf16*)alloc((size_t)HIDq * Cq * 2);
  __bf16* w2b  = (__bf16*)alloc((size_t)Cq * HIDq * 2);
  __bf16* qbuf = (__bf16*)alloc(BNC * 2);              // (b,h,n,d)
  __bf16* kbuf = (__bf16*)alloc(BNC * 2);              // (b,h,n,d)
  __bf16* vTb  = (__bf16*)alloc(BNC * 2);              // (b,h,d,n)
  float*  vsum = (float*)alloc((size_t)Bq * Hq * HDq * 4);
  __bf16* atto = (__bf16*)alloc(BNC * 2);              // attention out (b,n,c)
  float*  xres = (float*)alloc(BNC * 4);               // x + proj
  __bf16* h2b  = (__bf16*)alloc(BNC * 2);              // LN2 bf16
  __bf16* m1b  = (__bf16*)alloc((size_t)Bq * Nq * HIDq * 2);

  // 1) weight conversion fp32 -> bf16
  cvt_bf16_kernel<<<1024, 256, 0, stream>>>(wq, wqb, Cq * Cq);
  cvt_bf16_kernel<<<1024, 256, 0, stream>>>(wk, wkb, Cq * Cq);
  cvt_bf16_kernel<<<1024, 256, 0, stream>>>(wv, wvb, Cq * Cq);
  cvt_bf16_kernel<<<1024, 256, 0, stream>>>(w_proj, wpb, Cq * Cq);
  cvt_bf16_kernel<<<2048, 256, 0, stream>>>(w1, w1b, HIDq * Cq);
  cvt_bf16_kernel<<<2048, 256, 0, stream>>>(w2, w2b, Cq * HIDq);

  const int rows = Bq * Nq;             // 8192
  // 2) LN1
  ln_kernel<<<rows / 4, 128, 0, stream>>>(x_token, g1, b1, tb);

  // 3) QKV projections (each: 512x32 = 16384 tiles, 4 waves/block)
  const int qkvBlocks = (rows / 16) * (Cq / 16) / 4;   // 4096
  qkv_gemm_kernel<<<qkvBlocks, 128, 0, stream>>>(tb, wqb, qbuf, 0);
  qkv_gemm_kernel<<<qkvBlocks, 128, 0, stream>>>(tb, wkb, kbuf, 1);
  qkv_gemm_kernel<<<qkvBlocks, 128, 0, stream>>>(tb, wvb, vTb, 2);

  // 4) per-(b,h) V row sums
  vsum_kernel<<<Bq * Hq, HDq, 0, stream>>>(vTb, vsum);

  // 5) clustered attention: 64 (b,h) x 64 query tiles = 4096 waves
  attn_kernel<<<1024, 128, 0, stream>>>(qbuf, kbuf, vTb, vsum, idxc, atto);

  // 6) output proj + bias + residual -> xres (f32)
  gemm_kernel<<<qkvBlocks, 128, 0, stream>>>(atto, wpb, b_proj, x_token, xres,
                                             nullptr, rows, Cq, Cq, 1);
  // 7) LN2
  ln_kernel<<<rows / 4, 128, 0, stream>>>(xres, g2, b2, h2b);

  // 8) MLP up + exact GELU -> m1 (bf16)
  const int mlp1Blocks = (rows / 16) * (HIDq / 16) / 4;  // 16384
  gemm_kernel<<<mlp1Blocks, 128, 0, stream>>>(h2b, w1b, bb1, nullptr, nullptr,
                                              m1b, rows, Cq, HIDq, 2);
  // 9) MLP down + bias + residual -> final output (f32)
  gemm_kernel<<<qkvBlocks, 128, 0, stream>>>(m1b, w2b, bb2, xres,
                                             (float*)d_out, nullptr, rows, HIDq,
                                             Cq, 1);
}
